// UpsampleConvBlock_25975962206808
// MI455X (gfx1250) — matrix operand order repaired and checked by reading, not compiled, not run on hardware
//
#include <hip/hip_runtime.h>
#include <hip/hip_bf16.h>
#include <stddef.h>

// ---------------------------------------------------------------------------
// Problem constants (from reference): x(16,128,8192), W(128,384,3), b(128)
// out(16,128,16384) f32.
// ---------------------------------------------------------------------------
#define BATCH   16
#define CIN     128
#define LEN     8192
#define COUT    128
#define QPOW    3
#define CQ      (CIN*QPOW)          // 384 power-expanded channels
#define KDIM    (CQ*2)              // 768: 2 effective taps per parity
#define KT      (KDIM/32)           // 24 k-tiles of 32
#define MT      (COUT/16)           // 8 m-tiles of 16
#define NBLK    64                  // original positions per workgroup
#define PITCH   392                 // bf16 elems per LDS row (384 + 8 pad, 16B-mult)
#define L2LEN   (2*LEN)             // 16384 output positions per (b,c)

// Workspace layout (bytes)
#define WFRAG_OFF  0u                         // 2*8*24*512 bf16 = 384 KB weight fragments
#define WFRAG_ELEMS (2*MT*KT*512)             // 196608
#define GSUM_OFF   ((size_t)WFRAG_ELEMS*2)    // 393216
#define GSQ_OFF    (GSUM_OFF + 2048*4)        // +8 KB
#define OC_OFF     (GSQ_OFF + 2048*4)         // conv output planes [b][co][parity][L] f32 = 128 MB

typedef __attribute__((ext_vector_type(16))) __bf16    v16bf;
typedef __attribute__((ext_vector_type(8)))  float     v8f;
typedef __attribute__((ext_vector_type(4)))  unsigned int u32x4;

struct Frag32 { u32x4 lo, hi; };   // 32 bytes, trivially copyable -> bit_cast to v16bf

__device__ __forceinline__ unsigned short f2bf(float f) {
    unsigned int u = __builtin_bit_cast(unsigned int, f);
    unsigned int r = u + 0x7FFFu + ((u >> 16) & 1u);   // round-to-nearest-even
    return (unsigned short)(r >> 16);
}

// ---------------------------------------------------------------------------
// Kernel 0: zero InstanceNorm accumulators (sum + sumsq, 4096 floats)
// ---------------------------------------------------------------------------
__global__ void zero_stats_kernel(float* __restrict__ p, int n) {
    int i = blockIdx.x * blockDim.x + threadIdx.x;
    if (i < n) p[i] = 0.0f;
}

// ---------------------------------------------------------------------------
// Kernel 1: fold upsample into weights and swizzle into a contiguous-k 16-bit
// A-matrix (16x32) WMMA lane layout.  Storage index == linear tid:
//   (((p*8+mt)*24+kt)*32 + lane)*16 + e
// Lane/element -> (M,K):  M = lane%16 ;  K = e + (lane>=16 ? 16 : 0)
// K-dim ordering: Kg = tap*384 + cq.
// ---------------------------------------------------------------------------
__global__ void prep_weights_kernel(const float* __restrict__ W,
                                    unsigned short* __restrict__ wf) {
    int tid = blockIdx.x * blockDim.x + threadIdx.x;
    if (tid >= WFRAG_ELEMS) return;
    int e    = tid & 15;   int idx = tid >> 4;
    int lane = idx & 31;   idx >>= 5;
    int kt   = idx % 24;   idx /= 24;
    int mt   = idx & 7;    int p = idx >> 3;

    int co = mt * 16 + (lane & 15);
    int K  = e + ((lane & 16) ? 16 : 0);
    int Kg = kt * 32 + K;
    int t  = (Kg >= CQ) ? 1 : 0;
    int cq = Kg - t * CQ;

    const float* wr = W + ((size_t)co * CQ + cq) * 3;
    float w0 = wr[0], w1 = wr[1], w2 = wr[2];
    // even parity taps: (W0, W1+W2) on (x[i-1], x[i])
    // odd  parity taps: (W0+W1, W2) on (x[i],   x[i+1])
    float v = p ? (t ? w2 : (w0 + w1)) : (t ? (w1 + w2) : w0);
    wf[tid] = f2bf(v);
}

// ---------------------------------------------------------------------------
// Kernel 2: fused power-expansion + dual-parity conv as bf16 WMMA GEMM.
// Grid: (LEN/NBLK, BATCH).  256 threads = 8 waves.
//   wave -> parity = wave&1, m-tile pair = (wave>>1)*2 .. +1
// Each wave: 2 mtiles x 4 ntiles of 16x16 f32 accumulators (64 VGPRs).
// K-loop is split into two ROLLED halves (tap 0: kt 0..11, tap 1: kt 12..23)
// so the parity/tap position shift is loop-invariant and the LDS B offset is
// a pure +32 induction.  Accumulators stay loop-carried PHIs -> in-place
// WMMA accumulation, no v_mov shuffles, no hazard NOPs in the hot loop.
// ---------------------------------------------------------------------------
__global__ __launch_bounds__(256) void onn_gemm_kernel(
        const float* __restrict__ X,
        const unsigned short* __restrict__ wf,
        float* __restrict__ Oc,
        float* __restrict__ gs,
        float* __restrict__ gq) {
    __shared__ __align__(16) unsigned short xp[(NBLK + 2) * PITCH];

    const int b   = blockIdx.y;
    const int n0  = blockIdx.x * NBLK;
    const int tid = threadIdx.x;

    // ---- stage x, x^2, x^3 as bf16 into LDS (positions n0-1 .. n0+NBLK) ----
    for (int i = tid; i < CIN * (NBLK + 2); i += 256) {
        int c  = i / (NBLK + 2);
        int pl = i - c * (NBLK + 2);
        int gp = n0 - 1 + pl;                       // conv zero-padding at edges
        float x = 0.0f;
        if (gp >= 0 && gp < LEN) x = X[((size_t)b * CIN + c) * LEN + gp];
        float x2 = x * x;
        xp[pl * PITCH + c]           = f2bf(x);
        xp[pl * PITCH + CIN + c]     = f2bf(x2);
        xp[pl * PITCH + 2 * CIN + c] = f2bf(x2 * x);
    }
    __syncthreads();

    const int wave  = tid >> 5;
    const int lane  = tid & 31;
    const int p     = wave & 1;
    const int mt0   = (wave >> 1) * 2;
    const int lhalf = (lane >= 16) ? 1 : 0;
    const int l16   = lane & 15;

    // per-lane LDS row base (elements)
    const int rowbase = (l16 + 1) * PITCH + lhalf * 16;
    // A fragments for this wave start here; kt stride is 512 elements
    const size_t abase0 = ((size_t)((p * MT + mt0) * KT)) * 512 + lane * 16;

    v8f zero = {};
    v8f acc[2][4];
#pragma unroll
    for (int m = 0; m < 2; ++m)
#pragma unroll
        for (int n = 0; n < 4; ++n) acc[m][n] = zero;

    auto run_half = [&](int k0, int k1, int shift) {
        const int off_base = rowbase + shift * PITCH;  // loop-invariant
#pragma unroll 1
        for (int kt = k0; kt < k1; ++kt) {
            // A fragments: pre-swizzled weights, 32B/lane, coalesced, L2-resident
            const size_t aoff = abase0 + (size_t)kt * 512;
            const u32x4* ap0 = (const u32x4*)(wf + aoff);
            const u32x4* ap1 = (const u32x4*)(wf + aoff + (size_t)KT * 512);
            Frag32 fa0; fa0.lo = ap0[0]; fa0.hi = ap0[1];
            Frag32 fa1; fa1.lo = ap1[0]; fa1.hi = ap1[1];
            v16bf aF0 = __builtin_bit_cast(v16bf, fa0);
            v16bf aF1 = __builtin_bit_cast(v16bf, fa1);
            if (kt + 1 < KT) __builtin_prefetch(wf + aoff + 512, 0, 1);

            const int off0 = off_base + (kt - k0) * 32;   // cqb induction (+32)
#pragma unroll
            for (int nt = 0; nt < 4; ++nt) {
                // B fragment: 16 contiguous bf16 channels at this lane's position
                const u32x4* bp = (const u32x4*)&xp[off0 + nt * (16 * PITCH)];
                Frag32 fb; fb.lo = bp[0]; fb.hi = bp[1];  // 2x ds_load_b128
                v16bf bF = __builtin_bit_cast(v16bf, fb);

                acc[0][nt] = __builtin_amdgcn_wmma_f32_16x16x32_bf16(
                    false, aF0, false, bF, (short)0, acc[0][nt], false, false);
                acc[1][nt] = __builtin_amdgcn_wmma_f32_16x16x32_bf16(
                    false, aF1, false, bF, (short)0, acc[1][nt], false, false);
            }
        }
    };
    // tap 0 half: even parity reads x[i-1] (shift -1), odd reads x[i] (0)
    run_half(0, 12, p ? 0 : -1);
    // tap 1 half: even parity reads x[i] (0), odd reads x[i+1] (+1)
    run_half(12, 24, p ? 1 : 0);

    // ---- epilogue: conv planes [b][co][parity][L] + InstanceNorm partials ----
#pragma unroll
    for (int m = 0; m < 2; ++m) {
        int mt = mt0 + m;
#pragma unroll
        for (int r = 0; r < 8; ++r) {
            int co = mt * 16 + r + lhalf * 8;          // C-layout: M = r + half*8
            size_t base = (((size_t)(b * COUT + co)) * 2 + p) * LEN + n0;
            float sr = 0.0f, qr = 0.0f;
#pragma unroll
            for (int nt = 0; nt < 4; ++nt) {
                float v = acc[m][nt][r];
                Oc[base + nt * 16 + l16] = v;          // 64B coalesced per row seg
                sr += v; qr += v * v;
            }
            // butterfly over the 16-lane half that shares this co
            for (int o = 1; o < 16; o <<= 1) {
                sr += __shfl_xor(sr, o, 32);
                qr += __shfl_xor(qr, o, 32);
            }
            if (l16 == 0) {
                atomicAdd(&gs[b * COUT + co], sr);
                atomicAdd(&gq[b * COUT + co], qr);
            }
        }
    }
}

// ---------------------------------------------------------------------------
// Kernel 3: InstanceNorm (affine=False) + tanh, re-interleave parity planes.
// Bias is omitted on purpose: constant-per-channel shift cancels exactly in
// (out - mean), and var is translation invariant.
// ---------------------------------------------------------------------------
__global__ __launch_bounds__(256) void norm_tanh_kernel(
        const float* __restrict__ Oc,
        const float* __restrict__ gs,
        const float* __restrict__ gq,
        float* __restrict__ out) {
    const int row = blockIdx.x;                        // b*128 + co
    const float inv = 1.0f / (float)L2LEN;
    float mean = gs[row] * inv;
    float var  = gq[row] * inv - mean * mean;          // population var == jnp.var
    float rstd = rsqrtf(var + 1e-5f);

    const float* ev = Oc + ((size_t)row * 2) * LEN;
    const float* od = ev + LEN;
    float2* o2 = (float2*)out + (size_t)row * LEN;

    for (int i = threadIdx.x; i < LEN; i += 256) {
        float2 t;
        t.x = tanhf((ev[i] - mean) * rstd);
        t.y = tanhf((od[i] - mean) * rstd);
        o2[i] = t;                                     // coalesced 8B stores
    }
}

// ---------------------------------------------------------------------------
extern "C" void kernel_launch(void* const* d_in, const int* in_sizes, int n_in,
                              void* d_out, int out_size, void* d_ws, size_t ws_size,
                              hipStream_t stream) {
    const float* X = (const float*)d_in[0];            // (16,128,8192)
    const float* W = (const float*)d_in[1];            // (128,384,3)
    // d_in[2] (bias) mathematically cancelled by InstanceNorm -> unused.
    float* out = (float*)d_out;

    char* ws = (char*)d_ws;
    unsigned short* wfrag = (unsigned short*)(ws + WFRAG_OFF);
    float* gs = (float*)(ws + GSUM_OFF);
    float* gq = (float*)(ws + GSQ_OFF);
    float* Oc = (float*)(ws + OC_OFF);

    zero_stats_kernel<<<(4096 + 255) / 256, 256, 0, stream>>>(gs, 4096); // gs+gq contiguous
    prep_weights_kernel<<<(WFRAG_ELEMS + 255) / 256, 256, 0, stream>>>(W, wfrag);

    dim3 grid(LEN / NBLK, BATCH);                      // 128 x 16 workgroups
    onn_gemm_kernel<<<grid, 256, 0, stream>>>(X, wfrag, Oc, gs, gq);

    norm_tanh_kernel<<<BATCH * COUT, 256, 0, stream>>>(Oc, gs, gq, out);
}
